// MPNEncoder_69578470195850
// MI455X (gfx1250) — compile-verified, matrix operand order checked by reading, not run on hardware
//
#include <hip/hip_runtime.h>

typedef _Float16 h16 __attribute__((ext_vector_type(16)));
typedef _Float16 h8  __attribute__((ext_vector_type(8)));
typedef _Float16 h4  __attribute__((ext_vector_type(4)));
typedef float    f8  __attribute__((ext_vector_type(8)));

#define HP 320           // padded hidden dim (N tiles = 20, K chunks = 10)
#define MAXNB 8

// ---------------------------------------------------------------------------
// Weight prep: transpose [Kvalid x Nvalid] f32 -> [Npad x Kpad] f16, zero-pad.
// dst[n*Kpad + k] = src[k*Nvalid + n]
// ---------------------------------------------------------------------------
__global__ void mpn_transpose_pad_f16(const float* __restrict__ src,
                                      _Float16* __restrict__ dst,
                                      int Kvalid, int Nvalid, int Kpad, int total) {
  int t = blockIdx.x * blockDim.x + threadIdx.x;
  if (t >= total) return;
  int k = t % Kpad;
  int n = t / Kpad;
  float v = (k < Kvalid && n < Nvalid) ? src[(size_t)k * Nvalid + n] : 0.0f;
  dst[t] = (_Float16)v;
}

// ---------------------------------------------------------------------------
// Row-wise f32 -> f16 convert with K zero-padding: dst[r*Kpad+k] = src[r*Kvalid+k]
// ---------------------------------------------------------------------------
__global__ void mpn_convert_pad_f16(const float* __restrict__ src,
                                    _Float16* __restrict__ dst,
                                    int Kvalid, int Kpad, long total) {
  long t = (long)blockIdx.x * blockDim.x + threadIdx.x;
  if (t >= total) return;
  int k = (int)(t % Kpad);
  long r = t / Kpad;
  dst[t] = (k < Kvalid) ? (_Float16)src[r * Kvalid + k] : (_Float16)0.0f;
}

// ---------------------------------------------------------------------------
// WMMA GEMM: one wave computes MT vertical 16x16 tiles (MT*16 rows x 16 cols).
// B fragment loaded once per k-chunk, reused across MT A fragments.
// Epilogue: in-wave LDS transpose of each 16x16 tile -> coalesced b128 I/O.
// MODE 0: out_inp = f16(acc); out_msg = f16(relu(acc))
// MODE 1: out_msg = relu(f16(acc) + inp_add)
// MODE 2: out_msg = relu(f16(acc + bias[n]))     (bias applied pre-transpose)
// A: f16 [M x lda];  BT: f16 [Npad x ldb] (row n = col n of W, K-padded w/ 0)
// ---------------------------------------------------------------------------
template <int MODE, int KCH, int MT>
__global__ void __launch_bounds__(32)
mpn_gemm_wmma(const _Float16* __restrict__ A, int lda, int M,
              const _Float16* __restrict__ BT, int ldb,
              const _Float16* __restrict__ inp_add,
              const float* __restrict__ bias, int Nvalid,
              _Float16* __restrict__ out_msg, _Float16* __restrict__ out_inp,
              int ldo) {
  __shared__ _Float16 tile[16 * 16];   // 512 B, one wave per block

  const int lane = threadIdx.x & 31;
  const int halfsel = lane >> 4;
  const int lrow = lane & 15;
  const int mtile0 = blockIdx.x * MT;
  const int nbase = blockIdx.y * 16;
  const int n = nbase + lrow;

  const _Float16* brow = BT + (size_t)n * ldb;

  const _Float16* arow[MT];
#pragma unroll
  for (int mt = 0; mt < MT; ++mt) {
    int mrow = (mtile0 + mt) * 16 + lrow;
    if (mrow >= M) mrow = M - 1;   // clamp for ragged tail blocks
    arow[mt] = A + (size_t)mrow * lda;
  }

  f8 acc[MT];
#pragma unroll
  for (int mt = 0; mt < MT; ++mt)
    acc[mt] = (f8){0.f, 0.f, 0.f, 0.f, 0.f, 0.f, 0.f, 0.f};

#pragma unroll
  for (int kc = 0; kc < KCH; ++kc) {
    const int k0 = kc * 32 + halfsel * 8;
    h8 b0 = *(const h8*)(brow + k0);
    h8 b1 = *(const h8*)(brow + k0 + 16);
    h16 b = __builtin_shufflevector(b0, b1, 0, 1, 2, 3, 4, 5, 6, 7,
                                    8, 9, 10, 11, 12, 13, 14, 15);
#pragma unroll
    for (int mt = 0; mt < MT; ++mt) {
      h8 a0 = *(const h8*)(arow[mt] + k0);
      h8 a1 = *(const h8*)(arow[mt] + k0 + 16);
      h16 a = __builtin_shufflevector(a0, a1, 0, 1, 2, 3, 4, 5, 6, 7,
                                      8, 9, 10, 11, 12, 13, 14, 15);
      acc[mt] = __builtin_amdgcn_wmma_f32_16x16x32_f16(
          /*neg_a=*/false, a, /*neg_b=*/false, b,
          /*c_mod=*/(short)0, acc[mt], /*reuse_a=*/false, /*reuse_b=*/false);
    }
  }

  // per-column bias (pre-transpose lane layout is column-indexed)
  float bv = 0.0f;
  if (MODE == 2) bv = (n < Nvalid) ? bias[n] : 0.0f;

#pragma unroll
  for (int mt = 0; mt < MT; ++mt) {
    const int mbase = (mtile0 + mt) * 16;
    // 1) write this lane's column of the tile (rows r+8*halfsel) into LDS
#pragma unroll
    for (int r = 0; r < 8; ++r) {
      float v = acc[mt][r];
      if (MODE == 2) v += bv;
      tile[(r + 8 * halfsel) * 16 + lrow] = (_Float16)v;
    }
    // 2) read back one row-chunk per lane (DS ops are in-order within a wave)
    const int row = lrow;
    h8 vals = *(const h8*)&tile[row * 16 + 8 * halfsel];
    const int m = mbase + row;
    if (m < M) {
      size_t o = (size_t)m * ldo + nbase + 8 * halfsel;
      if (MODE == 0) {
        *(h8*)(out_inp + o) = vals;
        h8 rl;
#pragma unroll
        for (int i = 0; i < 8; ++i)
          rl[i] = (vals[i] > (_Float16)0.0f) ? vals[i] : (_Float16)0.0f;
        *(h8*)(out_msg + o) = rl;
      } else if (MODE == 1) {
        h8 ip = *(const h8*)(inp_add + o);
        h8 rl;
#pragma unroll
        for (int i = 0; i < 8; ++i) {
          _Float16 t = vals[i] + ip[i];
          rl[i] = (t > (_Float16)0.0f) ? t : (_Float16)0.0f;
        }
        *(h8*)(out_msg + o) = rl;
      } else {
        h8 rl;
#pragma unroll
        for (int i = 0; i < 8; ++i)
          rl[i] = (vals[i] > (_Float16)0.0f) ? vals[i] : (_Float16)0.0f;
        *(h8*)(out_msg + o) = rl;
      }
    }
  }
}

// ---------------------------------------------------------------------------
// a_message[atom][c] = sum_k message[a2b[atom][k]][c]   (f16 in/out)
// 4 atoms per 320-thread block; 4 cols (b64) per thread.
// ---------------------------------------------------------------------------
__global__ void mpn_atom_aggregate(const _Float16* __restrict__ msg,
                                   const int* __restrict__ a2b,
                                   _Float16* __restrict__ amsg,
                                   int max_nb, int H, int n_atoms) {
  const int sub = threadIdx.x / 80;      // atom within block
  const int chunk = threadIdx.x % 80;    // 4-col chunk
  const int atom = blockIdx.x * 4 + sub;
  const int c0 = chunk * 4;
  if (atom >= n_atoms || c0 >= H) return;
  int idx[MAXNB];
#pragma unroll
  for (int k = 0; k < MAXNB; ++k)
    idx[k] = (k < max_nb) ? a2b[(size_t)atom * max_nb + k] : 0;
  float s0 = 0.f, s1 = 0.f, s2 = 0.f, s3 = 0.f;
  for (int k = 0; k < max_nb; ++k) {
    h4 v = *(const h4*)(msg + (size_t)idx[k] * HP + c0);
    s0 += (float)v[0]; s1 += (float)v[1]; s2 += (float)v[2]; s3 += (float)v[3];
  }
  h4 o = { (_Float16)s0, (_Float16)s1, (_Float16)s2, (_Float16)s3 };
  *(h4*)(amsg + (size_t)atom * HP + c0) = o;
}

// ---------------------------------------------------------------------------
// m2[bond][c] = amsg[b2a[bond]][c] - msg[b2revb[bond]][c]
// 4 bonds per 320-thread block; 4 cols (b64) per thread.
// ---------------------------------------------------------------------------
__global__ void mpn_bond_update(const _Float16* __restrict__ msg,
                                const _Float16* __restrict__ amsg,
                                const int* __restrict__ b2a,
                                const int* __restrict__ b2revb,
                                _Float16* __restrict__ m2, int H, int n_bonds) {
  const int sub = threadIdx.x / 80;
  const int chunk = threadIdx.x % 80;
  const int bond = blockIdx.x * 4 + sub;
  const int c0 = chunk * 4;
  if (bond >= n_bonds || c0 >= H) return;
  const int a = b2a[bond];
  const int rb = b2revb[bond];
  h4 x = *(const h4*)(amsg + (size_t)a * HP + c0);
  h4 y = *(const h4*)(msg + (size_t)rb * HP + c0);
  h4 o;
#pragma unroll
  for (int i = 0; i < 4; ++i) o[i] = (_Float16)((float)x[i] - (float)y[i]);
  *(h4*)(m2 + (size_t)bond * HP + c0) = o;
}

// ---------------------------------------------------------------------------
// a_input[atom] = concat(f_atoms[atom] (f32->f16), sum_k msg[a2b[atom][k]]),
// zero-padded to Kpad. One block per atom.
// ---------------------------------------------------------------------------
__global__ void mpn_build_ainput(const float* __restrict__ f_atoms,
                                 const _Float16* __restrict__ msg,
                                 const int* __restrict__ a2b,
                                 _Float16* __restrict__ ain,
                                 int atom_fdim, int H, int max_nb, int Kpad) {
  int atom = blockIdx.x;
  int idx[MAXNB];
#pragma unroll
  for (int k = 0; k < MAXNB; ++k)
    idx[k] = (k < max_nb) ? a2b[(size_t)atom * max_nb + k] : 0;
  for (int c = threadIdx.x; c < Kpad; c += blockDim.x) {
    _Float16 v;
    if (c < atom_fdim) {
      v = (_Float16)f_atoms[(size_t)atom * atom_fdim + c];
    } else if (c < atom_fdim + H) {
      int cc = c - atom_fdim;
      float s = 0.f;
      for (int k = 0; k < max_nb; ++k)
        s += (float)msg[(size_t)idx[k] * HP + cc];
      v = (_Float16)s;
    } else {
      v = (_Float16)0.0f;
    }
    ain[(size_t)atom * Kpad + c] = v;
  }
}

// ---------------------------------------------------------------------------
// Deterministic ragged mean over sorted mol_id (binary search per molecule).
// Accumulates f16 atom_hiddens into f32, writes f32 mol_vecs.
// ---------------------------------------------------------------------------
__global__ void mpn_segment_mean(const _Float16* __restrict__ ah,
                                 const int* __restrict__ mol_id,
                                 int n_atoms, int H,
                                 float* __restrict__ out) {
  int m = blockIdx.x;
  int lo = 0, hi = n_atoms;
  while (lo < hi) { int mid = (lo + hi) >> 1; if (mol_id[mid] < m) lo = mid + 1; else hi = mid; }
  int start = lo;
  lo = start; hi = n_atoms;
  while (lo < hi) { int mid = (lo + hi) >> 1; if (mol_id[mid] < m + 1) lo = mid + 1; else hi = mid; }
  int end = lo;
  int cnt = end - start;
  int c = threadIdx.x;
  if (c >= H) return;
  float s = 0.f;
  for (int a = start; a < end; ++a) s += (float)ah[(size_t)a * HP + c];
  out[(size_t)m * H + c] = (cnt > 0) ? (s / (float)cnt) : 0.0f;
}

// ---------------------------------------------------------------------------
extern "C" void kernel_launch(void* const* d_in, const int* in_sizes, int n_in,
                              void* d_out, int out_size, void* d_ws, size_t ws_size,
                              hipStream_t stream) {
  const float* f_atoms = (const float*)d_in[0];
  const float* f_bonds = (const float*)d_in[1];
  const int*   a2b     = (const int*)d_in[2];
  const int*   b2a     = (const int*)d_in[3];
  const int*   b2revb  = (const int*)d_in[4];
  const int*   mol_id  = (const int*)d_in[5];
  const float* W_i     = (const float*)d_in[7];
  const float* W_h     = (const float*)d_in[8];
  const float* W_o     = (const float*)d_in[9];
  const float* b_o     = (const float*)d_in[10];

  const int H         = in_sizes[10];               // 300
  const int bond_fdim = in_sizes[7] / H;            // 147
  const int atom_fdim = in_sizes[9] / H - H;        // 133
  const int n_atoms   = in_sizes[0] / atom_fdim;    // 50000
  const int n_bonds   = in_sizes[1] / bond_fdim;    // 100000
  const int max_nb    = in_sizes[2] / n_atoms;      // 6
  const int n_mols    = out_size / H;               // 2000
  const int DEPTH     = 3;
  const int MT        = 4;                          // M tiles per wave

  const int Kp_i = ((bond_fdim + 31) / 32) * 32;    // 160  (5 chunks)
  const int Kin  = atom_fdim + H;                   // 433
  const int Kp_o = ((Kin + 31) / 32) * 32;          // 448  (14 chunks)

  // workspace carve-up (with buffer reuse)
  char* ws = (char*)d_ws;
  size_t off = 0;
  auto carve = [&](size_t bytes) -> char* {
    char* p = ws + off;
    off = (off + bytes + 255) & ~(size_t)255;
    return p;
  };
  size_t msg_bytes  = (size_t)n_bonds * HP * sizeof(_Float16);
  size_t ah_bytes   = (size_t)n_atoms * HP * sizeof(_Float16);   // f16 atom_hiddens
  size_t ain_bytes  = (size_t)n_atoms * Kp_o * sizeof(_Float16);
  size_t amsg_bytes = (size_t)n_atoms * HP * sizeof(_Float16);
  size_t fb16_bytes = (size_t)n_bonds * Kp_i * sizeof(_Float16);

  _Float16* msg  = (_Float16*)carve(msg_bytes);
  char*     inpR = carve(msg_bytes > ah_bytes ? msg_bytes : ah_bytes);   // inp f16 -> atom_hiddens f16
  char*     m2R  = carve(msg_bytes > ain_bytes ? msg_bytes : ain_bytes); // m2 f16 -> a_input f16
  char*     agR  = carve(amsg_bytes > fb16_bytes ? amsg_bytes : fb16_bytes); // fb16 -> amsg
  _Float16* WiT  = (_Float16*)carve((size_t)HP * Kp_i * sizeof(_Float16));
  _Float16* WhT  = (_Float16*)carve((size_t)HP * HP * sizeof(_Float16));
  _Float16* WoT  = (_Float16*)carve((size_t)HP * Kp_o * sizeof(_Float16));

  _Float16* inp  = (_Float16*)inpR;
  _Float16* ah   = (_Float16*)inpR;
  _Float16* m2   = (_Float16*)m2R;
  _Float16* ain  = (_Float16*)m2R;
  _Float16* fb16 = (_Float16*)agR;   // alive only until init GEMM completes
  _Float16* amsg = (_Float16*)agR;

  // --- weight prep (transposed, zero-padded, f16) ---
  {
    int tot = HP * Kp_i;
    mpn_transpose_pad_f16<<<(tot + 255) / 256, 256, 0, stream>>>(W_i, WiT, bond_fdim, H, Kp_i, tot);
    tot = HP * HP;
    mpn_transpose_pad_f16<<<(tot + 255) / 256, 256, 0, stream>>>(W_h, WhT, H, H, HP, tot);
    tot = HP * Kp_o;
    mpn_transpose_pad_f16<<<(tot + 255) / 256, 256, 0, stream>>>(W_o, WoT, Kin, H, Kp_o, tot);
  }

  // --- f_bonds -> f16, zero K-padding ---
  {
    long tot = (long)n_bonds * Kp_i;
    mpn_convert_pad_f16<<<(unsigned)((tot + 255) / 256), 256, 0, stream>>>(
        f_bonds, fb16, bond_fdim, Kp_i, tot);
  }

  const dim3 blk32(32, 1, 1);
  const int  mtiles_b = (n_bonds + 15) / 16;
  const int  mtiles_a = (n_atoms + 15) / 16;
  const dim3 grid_b((mtiles_b + MT - 1) / MT, HP / 16, 1);
  const dim3 grid_a((mtiles_a + MT - 1) / MT, HP / 16, 1);

  // --- inp = f_bonds @ W_i ; message = relu(inp) ---
  mpn_gemm_wmma<0, 5, 4><<<grid_b, blk32, 0, stream>>>(
      fb16, Kp_i, n_bonds, WiT, Kp_i, nullptr, nullptr, H,
      msg, inp, HP);

  // --- message passing iterations ---
  for (int it = 0; it < DEPTH - 1; ++it) {
    mpn_atom_aggregate<<<(n_atoms + 3) / 4, 320, 0, stream>>>(msg, a2b, amsg, max_nb, H, n_atoms);
    mpn_bond_update<<<(n_bonds + 3) / 4, 320, 0, stream>>>(msg, amsg, b2a, b2revb, m2, H, n_bonds);
    mpn_gemm_wmma<1, 10, 4><<<grid_b, blk32, 0, stream>>>(
        m2, HP, n_bonds, WhT, HP, inp, nullptr, H,
        msg, nullptr, HP);
  }

  // --- readout GEMM: atom_hiddens = relu(a_input @ W_o + b_o), f16 out ---
  mpn_build_ainput<<<n_atoms, 256, 0, stream>>>(f_atoms, msg, a2b, ain, atom_fdim, H, max_nb, Kp_o);
  mpn_gemm_wmma<2, 14, 4><<<grid_a, blk32, 0, stream>>>(
      ain, Kp_o, n_atoms, WoT, Kp_o, nullptr, b_o, H,
      ah, nullptr, HP);

  // --- per-molecule mean (deterministic, sorted mol_id) ---
  mpn_segment_mean<<<n_mols, HP, 0, stream>>>(ah, mol_id, n_atoms, H, (float*)d_out);
}
